// Decoder_40759239639201
// MI455X (gfx1250) — matrix operand order repaired and checked by reading, not compiled, not run on hardware
//
#include <hip/hip_runtime.h>
#include <hip/hip_bf16.h>

// ---------------- constants ----------------
#define BB   32      // batch
#define SS   512     // encoder length
#define TT   500     // decoder steps
#define MEL  80
#define PRE  256
#define ENC  512
#define DEC  1024
#define ATT  128
#define LCH  32
#define LK   31
#define POST 512
#define PK   5
#define TP   504     // padded time length (2 halo each side)

typedef __attribute__((ext_vector_type(16))) __bf16 v16bf;
typedef __attribute__((ext_vector_type(8)))  float  v8f;
typedef unsigned short u16;

union ABPack { v16bf v; unsigned u[8]; };

__device__ __forceinline__ u16 f2bf(float f) {
    unsigned x = __float_as_uint(f);
    x += 0x7fffu + ((x >> 16) & 1u);            // RNE
    return (u16)(x >> 16);
}
__device__ __forceinline__ unsigned pk2(float a, float b) {
    return (unsigned)f2bf(a) | ((unsigned)f2bf(b) << 16);
}
__device__ __forceinline__ float sigm(float x) { return 1.f / (1.f + __expf(-x)); }

// ---- CDNA5 async global->LDS copy (ASYNCcnt path) ----
__device__ __forceinline__ void async_b128(void* lds, const void* g) {
    unsigned l = (unsigned)(uintptr_t)lds;                 // flat low 32 bits == LDS offset
    unsigned long long ga = (unsigned long long)(uintptr_t)g;
    asm volatile("global_load_async_to_lds_b128 %0, %1, off" :: "v"(l), "v"(ga) : "memory");
}
__device__ __forceinline__ void async_wait0() {
    asm volatile("s_wait_asynccnt 0x0" ::: "memory");
}

// ---------------- WMMA core ----------------
// B packed layout: [K/32][N][16] u32; lane ln16 = N, half-wave = K+8 (in each 16), u[j] = K pair.
__device__ __forceinline__ void load_b(ABPack& b, const unsigned* __restrict__ Bp,
                                       int N, int col0, int k0, int half, int ln16) {
    const uint4* bq = (const uint4*)(Bp + ((size_t)(k0 >> 5) * N + col0 + ln16) * 16 + half * 4);
    uint4 b0 = bq[0], b1 = bq[2];
    b.u[0] = b0.x; b.u[1] = b0.y; b.u[2] = b0.z; b.u[3] = b0.w;
    b.u[4] = b1.x; b.u[5] = b1.y; b.u[6] = b1.z; b.u[7] = b1.w;
}

// A: bf16 row (pair-packed by construction); per chunk = two b128 loads.
__device__ __forceinline__ void load_a(ABPack& a, const u16* __restrict__ row, int k0, int half) {
    const unsigned* r = (const unsigned*)(row + k0);       // pair view (k0 even)
    const uint4* q0 = (const uint4*)(r + half * 4);
    const uint4* q1 = (const uint4*)(r + 8 + half * 4);
    uint4 x = *q0, y = *q1;
    a.u[0] = x.x; a.u[1] = x.y; a.u[2] = x.z; a.u[3] = x.w;
    a.u[4] = y.x; a.u[5] = y.y; a.u[6] = y.z; a.u[7] = y.w;
}

// Direct A: two bf16 row pointers (split at ksplit; ksplit multiple of 32).
__device__ __forceinline__ v8f wmma_tile_split(const u16* __restrict__ rowA,
                                               const u16* __restrict__ rowB, int ksplit,
                                               const unsigned* __restrict__ Bp,
                                               int N, int col0, int K) {
    const int lane = threadIdx.x & 31;
    const int half = lane >> 4;
    const int ln16 = lane & 15;
    v8f acc = {};
    for (int k0 = 0; k0 < K; k0 += 32) {
        const u16* r = (k0 < ksplit) ? (rowA + k0) : (rowB + (k0 - ksplit));
        ABPack a, b;
        load_a(a, r, 0, half);
        load_b(b, Bp, N, col0, k0, half, ln16);
        acc = __builtin_amdgcn_wmma_f32_16x16x32_bf16(false, a.v, false, b.v,
                                                      (short)0, acc, false, false);
    }
    return acc;
}

// Functor A (bf16 gather) for postnet im2col.
template <class AF>
__device__ __forceinline__ v8f wmma_tile_f(const AF& afetch, const unsigned* __restrict__ Bp,
                                           int N, int col0, int K) {
    const int lane = threadIdx.x & 31;
    const int half = lane >> 4;
    const int ln16 = lane & 15;
    v8f acc = {};
    for (int k0 = 0; k0 < K; k0 += 32) {
        ABPack a, b;
        load_b(b, Bp, N, col0, k0, half, ln16);
#pragma unroll
        for (int j = 0; j < 8; ++j) {
            const int ka = k0 + (j < 4 ? 0 : 16) + half * 8 + (j & 3) * 2;
            a.u[j] = (unsigned)afetch(ka) | ((unsigned)afetch(ka + 1) << 16);
        }
        acc = __builtin_amdgcn_wmma_f32_16x16x32_bf16(false, a.v, false, b.v,
                                                      (short)0, acc, false, false);
    }
    return acc;
}

template <class EP>
__device__ __forceinline__ void tile_store(v8f acc, const EP& ep) {
    const int lane = threadIdx.x & 31;
    const int half = lane >> 4;
    const int ln16 = lane & 15;
#pragma unroll
    for (int j = 0; j < 8; ++j) ep(j + 8 * half, ln16, acc[j]);
}

// ---------------- weight packer: fp32 W(N,K) -> u32 Bp[Kpad/32][N][16] ----------
__global__ void k_pack_weight(const float* __restrict__ W, unsigned* __restrict__ Bp,
                              int N, int K, int Kpad) {
    int idx = blockIdx.x * blockDim.x + threadIdx.x;
    int total = (Kpad >> 5) * N * 16;
    if (idx >= total) return;
    int c = idx / (N * 16);
    int rem = idx - c * N * 16;
    int n = rem >> 4, p = rem & 15;
    int k = c * 32 + p * 2;
    float a = (k     < K) ? W[(size_t)n * K + k]     : 0.f;
    float b = (k + 1 < K) ? W[(size_t)n * K + k + 1] : 0.f;
    Bp[idx] = pk2(a, b);
}

__global__ void k_zero(float* p, int n) {
    int i = blockIdx.x * blockDim.x + threadIdx.x;
    if (i < n) p[i] = 0.f;
}

__global__ void k_f2b(const float* __restrict__ in, u16* __restrict__ out, int n) {
    int i = blockIdx.x * blockDim.x + threadIdx.x;
    if (i < n) out[i] = f2bf(in[i]);
}

// ---------------- build teacher-forced prenet input DINB (T*B, 96) bf16 ----------------
__global__ void k_din(const float* __restrict__ melt, u16* __restrict__ din) {
    int idx = blockIdx.x * blockDim.x + threadIdx.x;
    if (idx >= TT * BB * 96) return;
    int m = idx / 96, k = idx - m * 96;
    int t = m >> 5, b = m & 31;                 // m = t*32 + b
    float v = 0.f;
    if (k < MEL && t > 0) v = melt[((size_t)b * MEL + k) * TT + (t - 1)];
    din[idx] = f2bf(v);
}

// ---------------- prenet ----------------
__global__ void k_prenet1(const u16* __restrict__ din, const unsigned* __restrict__ W1P,
                          const float* __restrict__ b1, u16* __restrict__ P1B) {
    int wave = threadIdx.x >> 5;
    int tile = blockIdx.x * (blockDim.x >> 5) + wave;  // 1000 * 16 tiles
    int mt = tile >> 4, nt = tile & 15;
    int row0 = mt * 16, col0 = nt * 16;
    int ln16 = threadIdx.x & 15;
    const u16* rowA = din + (size_t)(row0 + ln16) * 96;
    v8f acc = wmma_tile_split(rowA, rowA, 96, W1P, PRE, col0, 96);
    tile_store(acc, [&](int mi, int ni, float v) {
        int m = row0 + mi, n = col0 + ni;
        P1B[(size_t)m * PRE + n] = f2bf(fmaxf(v + b1[n], 0.f));
    });
}

__global__ void k_prenet2(const u16* __restrict__ P1B, const unsigned* __restrict__ W2P,
                          const float* __restrict__ b2, u16* __restrict__ PB) {
    int wave = threadIdx.x >> 5;
    int tile = blockIdx.x * (blockDim.x >> 5) + wave;
    int mt = tile >> 4, nt = tile & 15;
    int row0 = mt * 16, col0 = nt * 16;
    int ln16 = threadIdx.x & 15;
    const u16* rowA = P1B + (size_t)(row0 + ln16) * PRE;
    v8f acc = wmma_tile_split(rowA, rowA, PRE, W2P, PRE, col0, PRE);
    tile_store(acc, [&](int mi, int ni, float v) {
        PB[(size_t)(row0 + mi) * PRE + col0 + ni] = f2bf(fmaxf(v + b2[col0 + ni], 0.f));
    });
}

// ---------------- processed_mem: (B*S,512)bf16 @ WmP -> pm fp32 ----------------
__global__ void k_pm(const u16* __restrict__ encB, const unsigned* __restrict__ WmP,
                     float* __restrict__ pm) {
    int wave = threadIdx.x >> 5;
    int tile = blockIdx.x * (blockDim.x >> 5) + wave;  // 1024 * 8 tiles
    int mt = tile >> 3, nt = tile & 7;
    int row0 = mt * 16, col0 = nt * 16;
    int ln16 = threadIdx.x & 15;
    const u16* rowA = encB + (size_t)(row0 + ln16) * ENC;
    v8f acc = wmma_tile_split(rowA, rowA, ENC, WmP, ATT, col0, ENC);
    tile_store(acc, [&](int mi, int ni, float v) {
        pm[(size_t)(row0 + mi) * ATT + col0 + ni] = v;
    });
}

// ---------------- per-step recurrent GEMMs ----------------
__global__ void k_recurrent(const u16* __restrict__ h1B, const u16* __restrict__ h2B,
                            const unsigned* __restrict__ WqP, const unsigned* __restrict__ Whh1P,
                            const unsigned* __restrict__ Whh2P,
                            float* __restrict__ q, float* __restrict__ gh1, float* __restrict__ gh2) {
    int wave = threadIdx.x >> 5;
    int tile = blockIdx.x * (blockDim.x >> 5) + wave;  // 16 + 512 + 512 = 1040 tiles
    const u16* A; const unsigned* Bp; float* out; int N, id;
    if (tile < 16)        { A = h2B; Bp = WqP;   out = q;   N = ATT;     id = tile; }
    else if (tile < 528)  { A = h1B; Bp = Whh1P; out = gh1; N = 4 * DEC; id = tile - 16; }
    else                  { A = h2B; Bp = Whh2P; out = gh2; N = 4 * DEC; id = tile - 528; }
    int ntn = N >> 4;
    int mt = id / ntn, nt = id - mt * ntn;
    int row0 = mt * 16, col0 = nt * 16;
    int ln16 = threadIdx.x & 15;
    const u16* rowA = A + (size_t)(row0 + ln16) * DEC;
    v8f acc = wmma_tile_split(rowA, rowA, DEC, Bp, N, col0, DEC);
    tile_store(acc, [&](int mi, int ni, float v) {
        out[(size_t)(row0 + mi) * N + col0 + ni] = v;
    });
}

// ---------------- attention (async-LDS staging + VALU math) ----------------
__global__ void k_attention(const float* __restrict__ prev_attn, const float* __restrict__ q,
                            const float* __restrict__ pm, const float* __restrict__ Wconv,
                            const float* __restrict__ Wloc, const float* __restrict__ vvec,
                            const float* __restrict__ enc, float* __restrict__ attn_dout,
                            float* __restrict__ attn_cur, float* __restrict__ ctx,
                            u16* __restrict__ ctxB, int t) {
    int b = blockIdx.x;
    int s = threadIdx.x;                       // 512 threads
    __shared__ __align__(16) float pa[SS + 32];
    __shared__ __align__(16) float wc[LCH * LK];    // 992
    __shared__ __align__(16) float wl[ATT * LCH];   // 4096
    __shared__ __align__(16) float qs[ATT];
    __shared__ float red[SS];
    if (s < 16) { pa[s] = 0.f; pa[SS + 16 + s] = 0.f; }
    // CDNA5 async global->LDS staging (b128 per lane), tracked by ASYNCcnt
    if (s < 128) async_b128(&pa[16 + s * 4], prev_attn + (size_t)b * SS + s * 4);
    if (s < 248) async_b128(&wc[s * 4], Wconv + s * 4);
    if (s < 32)  async_b128(&qs[s * 4], q + (size_t)b * ATT + s * 4);
    async_b128(&wl[s * 4], Wloc + s * 4);
    async_b128(&wl[(s + 512) * 4], Wloc + (size_t)(s + 512) * 4);
    async_wait0();
    __syncthreads();

    float conv[LCH];
#pragma unroll
    for (int c = 0; c < LCH; ++c) {
        float acc = 0.f;
#pragma unroll
        for (int k = 0; k < LK; ++k) acc += wc[c * LK + k] * pa[s + k + 1];  // s+k-15+16
        conv[c] = acc;
    }
    const float* pmrow = pm + ((size_t)b * SS + s) * ATT;
    float e = 0.f;
    for (int a = 0; a < ATT; ++a) {
        float loc = 0.f;
#pragma unroll
        for (int c = 0; c < LCH; ++c) loc += wl[a * LCH + c] * conv[c];
        e += vvec[a] * tanhf(qs[a] + pmrow[a] + loc);
    }
    // softmax over S
    red[s] = e; __syncthreads();
    for (int off = 256; off > 0; off >>= 1) { if (s < off) red[s] = fmaxf(red[s], red[s + off]); __syncthreads(); }
    float mx = red[0]; __syncthreads();
    float ex = __expf(e - mx);
    red[s] = ex; __syncthreads();
    for (int off = 256; off > 0; off >>= 1) { if (s < off) red[s] += red[s + off]; __syncthreads(); }
    float aw = ex / red[0];
    __syncthreads();
    red[s] = aw;
    attn_cur[(size_t)b * SS + s] = aw;
    attn_dout[((size_t)b * TT + t) * SS + s] = aw;
    __syncthreads();
    // context: thread s computes channel e = s
    float cacc = 0.f;
    const float* eb = enc + (size_t)b * SS * ENC;
    for (int ss = 0; ss < SS; ++ss) cacc += red[ss] * eb[(size_t)ss * ENC + s];
    ctx[(size_t)b * ENC + s] = cacc;
    ctxB[(size_t)b * ENC + s] = f2bf(cacc);
}

// ---------------- fused LSTM cell ----------------
__global__ void k_lstm(const u16* __restrict__ src0, int ld0, const u16* __restrict__ src1,
                       int ld1, int ksplit, int K, const unsigned* __restrict__ WP,
                       const float* __restrict__ gh, const float* __restrict__ bih,
                       const float* __restrict__ bhh, float* __restrict__ h,
                       u16* __restrict__ hB, float* __restrict__ c) {
    __shared__ float gl[4][32][16];
    int wave = threadIdx.x >> 5;               // 8 waves
    int gi = wave >> 1, mt = wave & 1;
    int col0g = blockIdx.x * 16;               // 64 blocks cover 1024 per-gate columns
    int colN = gi * DEC + col0g;
    int row0 = mt * 16;
    int ln16 = threadIdx.x & 15;
    const u16* rowA = src0 + (size_t)(row0 + ln16) * ld0;
    const u16* rowB = src1 + (size_t)(row0 + ln16) * ld1;
    v8f acc = wmma_tile_split(rowA, rowB, ksplit, WP, 4 * DEC, colN, K);
    tile_store(acc, [&](int mi, int ni, float v) {
        int m = row0 + mi, n = colN + ni;
        gl[gi][m][ni] = v + gh[(size_t)m * 4 * DEC + n] + bih[n] + bhh[n];
    });
    __syncthreads();
    for (int idx = threadIdx.x; idx < 512; idx += blockDim.x) {
        int m = idx >> 4, ni = idx & 15;
        int n = col0g + ni;
        float ig = gl[0][m][ni], fg = gl[1][m][ni], gg = gl[2][m][ni], og = gl[3][m][ni];
        float cp = c[(size_t)m * DEC + n];
        float cn = sigm(fg) * cp + sigm(ig) * tanhf(gg);
        float hn = sigm(og) * tanhf(cn);
        c[(size_t)m * DEC + n] = cn;
        h[(size_t)m * DEC + n] = hn;
        hB[(size_t)m * DEC + n] = f2bf(hn);
    }
}

// ---------------- mel + stop projection ----------------
__global__ void k_proj(const u16* __restrict__ h2B, const u16* __restrict__ ctxB,
                       const float* __restrict__ h2, const float* __restrict__ ctx,
                       const unsigned* __restrict__ WmelP, const float* __restrict__ bmel,
                       const float* __restrict__ Wstop, const float* __restrict__ bstop,
                       float* __restrict__ mel_dout, float* __restrict__ stop_dout, int t) {
    int wave = threadIdx.x >> 5;               // 12 waves
    if (wave < 10) {
        int tm = wave / 5, tn = wave - tm * 5;
        int row0 = tm * 16, col0 = tn * 16;
        int ln16 = threadIdx.x & 15;
        const u16* rowA = h2B + (size_t)(row0 + ln16) * DEC;
        const u16* rowB = ctxB + (size_t)(row0 + ln16) * ENC;
        v8f acc = wmma_tile_split(rowA, rowB, DEC, WmelP, MEL, col0, DEC + ENC);
        tile_store(acc, [&](int mi, int ni, float v) {
            int b = row0 + mi, n = col0 + ni;
            mel_dout[((size_t)b * MEL + n) * TT + t] = v + bmel[n];
        });
    } else if (wave == 10) {
        int b = threadIdx.x & 31;
        float acc = bstop[0];
        for (int k = 0; k < DEC; ++k) acc += h2[(size_t)b * DEC + k] * Wstop[k];
        for (int k = 0; k < ENC; ++k) acc += ctx[(size_t)b * ENC + k] * Wstop[DEC + k];
        stop_dout[(size_t)b * TT + t] = sigm(acc);
    }
}

// ---------------- pad-fill: out_mel (B,80,500) fp32 -> Xp0 (B,84,504) bf16 --------------
__global__ void k_padfill(const float* __restrict__ src, u16* __restrict__ Xp) {
    int idx = blockIdx.x * blockDim.x + threadIdx.x;
    if (idx >= BB * MEL * TT) return;
    int b = idx / (MEL * TT);
    int rem = idx - b * MEL * TT;
    int c = rem / TT, l = rem - c * TT;
    Xp[((size_t)b * 84 + c) * TP + 2 + l] = f2bf(src[idx]);
}

// ---------------- postnet conv as im2col GEMM over padded bf16 input ---------------------
__global__ void k_postconv(const u16* __restrict__ Xp, int Cpad,
                           const unsigned* __restrict__ WP, int N, int Kpad,
                           const float* __restrict__ bias, const float* __restrict__ g,
                           const float* __restrict__ beta, const float* __restrict__ mean,
                           const float* __restrict__ var, int mode,
                           const float* __restrict__ resid, u16* __restrict__ Ypad,
                           float* __restrict__ Yout) {
    int wave = threadIdx.x >> 5;
    int ntn = N >> 4;
    int tile = blockIdx.x * (blockDim.x >> 5) + wave;
    int mt = tile / ntn, nt = tile - mt * ntn;
    if (mt >= (BB * TT) / 16) return;
    int row0 = mt * 16, col0 = nt * 16;
    int ln16 = threadIdx.x & 15;
    int m = row0 + ln16;
    int b = m / TT, l = m - b * TT;
    const u16* rowbase = Xp + (size_t)b * Cpad * TP + l;   // x[ci][l+kk-2] -> rowbase[ci*TP+kk]
    auto af = [&](int k) -> u16 {
        int ci = k / PK, kk = k - ci * PK;
        return rowbase[(size_t)ci * TP + kk];
    };
    v8f acc = wmma_tile_f(af, WP, N, col0, Kpad);
    tile_store(acc, [&](int mi, int ni, float v) {
        int mm = row0 + mi, n = col0 + ni;
        int bb = mm / TT, ll = mm - bb * TT;
        float y = v + bias[n];
        if (mode == 0) {
            y = tanhf(g[n] * (y - mean[n]) * rsqrtf(var[n] + 1e-5f) + beta[n]);
            Ypad[((size_t)bb * N + n) * TP + 2 + ll] = f2bf(y);
        } else {
            size_t oi = ((size_t)bb * N + n) * TT + ll;
            Yout[oi] = y + resid[oi];
        }
    });
}

// ---------------- host driver -------------------------------------------------------------
extern "C" void kernel_launch(void* const* d_in, const int* in_sizes, int n_in,
                              void* d_out, int out_size, void* d_ws, size_t ws_size,
                              hipStream_t stream) {
    (void)in_sizes; (void)n_in; (void)out_size; (void)ws_size;
    const float* enc   = (const float*)d_in[0];
    const float* melt  = (const float*)d_in[1];
    const float* W1    = (const float*)d_in[2];
    const float* b1    = (const float*)d_in[3];
    const float* W2    = (const float*)d_in[4];
    const float* b2    = (const float*)d_in[5];
    const float* Wq    = (const float*)d_in[6];
    const float* Wm    = (const float*)d_in[7];
    const float* Wconv = (const float*)d_in[8];
    const float* Wloc  = (const float*)d_in[9];
    const float* vvec  = (const float*)d_in[10];
    const float* Wih1  = (const float*)d_in[11];
    const float* Whh1  = (const float*)d_in[12];
    const float* bih1  = (const float*)d_in[13];
    const float* bhh1  = (const float*)d_in[14];
    const float* Wih2  = (const float*)d_in[15];
    const float* Whh2  = (const float*)d_in[16];
    const float* bih2  = (const float*)d_in[17];
    const float* bhh2  = (const float*)d_in[18];
    const float* Wmel  = (const float*)d_in[19];
    const float* bmel  = (const float*)d_in[20];
    const float* Wstop = (const float*)d_in[21];
    const float* bstop = (const float*)d_in[22];
    const float* pw[5], *pb[5], *pg[4], *pbe[4], *pmn[4], *pvr[4];
    for (int l = 0; l < 4; ++l) {
        pw[l]  = (const float*)d_in[23 + 6 * l + 0];
        pb[l]  = (const float*)d_in[23 + 6 * l + 1];
        pg[l]  = (const float*)d_in[23 + 6 * l + 2];
        pbe[l] = (const float*)d_in[23 + 6 * l + 3];
        pmn[l] = (const float*)d_in[23 + 6 * l + 4];
        pvr[l] = (const float*)d_in[23 + 6 * l + 5];
    }
    pw[4] = (const float*)d_in[47];
    pb[4] = (const float*)d_in[48];

    float* out_mel  = (float*)d_out;
    float* out_post = out_mel + (size_t)BB * MEL * TT;
    float* out_attn = out_post + (size_t)BB * MEL * TT;
    float* out_stop = out_attn + (size_t)BB * TT * SS;

    // workspace carve
    char* ws = (char*)d_ws;
    size_t cur = 0;
    auto carve = [&](size_t bytes) -> void* {
        void* p = ws + cur;
        cur = (cur + bytes + 255) & ~(size_t)255;
        return p;
    };
    unsigned* Wih1P = (unsigned*)carve((size_t)384 * 4096 * 4);
    unsigned* Whh1P = (unsigned*)carve((size_t)512 * 4096 * 4);
    unsigned* Wih2P = (unsigned*)carve((size_t)512 * 4096 * 4);
    unsigned* Whh2P = (unsigned*)carve((size_t)512 * 4096 * 4);
    unsigned* WqP   = (unsigned*)carve((size_t)512 * 128 * 4);
    unsigned* WmP   = (unsigned*)carve((size_t)256 * 128 * 4);
    unsigned* W1P   = (unsigned*)carve((size_t)48 * 256 * 4);
    unsigned* W2P   = (unsigned*)carve((size_t)128 * 256 * 4);
    unsigned* WmelP = (unsigned*)carve((size_t)768 * 80 * 4);
    unsigned* PP[5];
    PP[0] = (unsigned*)carve((size_t)208 * 512 * 4);
    PP[1] = (unsigned*)carve((size_t)1280 * 512 * 4);
    PP[2] = (unsigned*)carve((size_t)1280 * 512 * 4);
    PP[3] = (unsigned*)carve((size_t)1280 * 512 * 4);
    PP[4] = (unsigned*)carve((size_t)1280 * 80 * 4);
    u16* encB = (u16*)carve((size_t)BB * SS * ENC * 2);
    u16* DINB = (u16*)carve((size_t)TT * BB * 96 * 2);
    u16* P1B  = (u16*)carve((size_t)TT * BB * PRE * 2);
    u16* PB   = (u16*)carve((size_t)TT * BB * PRE * 2);
    float* pmem = (float*)carve((size_t)BB * SS * ATT * 4);
    float* zreg = (float*)carve((size_t)(4 * BB * DEC + BB * SS) * 4);  // h1,c1,h2,c2,attnZ
    float* h1 = zreg, *c1 = h1 + BB * DEC, *h2 = c1 + BB * DEC, *c2 = h2 + BB * DEC;
    float* attnZ = c2 + BB * DEC;
    u16* hBreg = (u16*)carve((size_t)2 * BB * DEC * 2);                 // h1B, h2B
    u16* h1B = hBreg, *h2B = h1B + BB * DEC;
    float* q    = (float*)carve((size_t)BB * ATT * 4);
    float* gh1  = (float*)carve((size_t)BB * 4 * DEC * 4);
    float* gh2  = (float*)carve((size_t)BB * 4 * DEC * 4);
    float* ctx  = (float*)carve((size_t)BB * ENC * 4);
    u16*   ctxB = (u16*)carve((size_t)BB * ENC * 2);
    float* attnA = (float*)carve((size_t)BB * SS * 4);
    float* attnB = (float*)carve((size_t)BB * SS * 4);
    size_t xp0n = (size_t)BB * 84 * TP;
    size_t xpn  = (size_t)BB * POST * TP;
    u16* xpall = (u16*)carve((xp0n + 2 * xpn) * 2);
    u16* Xp0 = xpall;
    u16* XpA = Xp0 + xp0n;
    u16* XpB = XpA + xpn;

    // ---- pack weights to bf16 WMMA-B layout ----
#define PACKW(Wp, Pp, Np, Kp, Kpadp) \
    k_pack_weight<<<(((Kpadp) / 2) * (Np) + 255) / 256, 256, 0, stream>>>(Wp, Pp, Np, Kp, Kpadp)
    PACKW(Wih1, Wih1P, 4096, 768, 768);
    PACKW(Whh1, Whh1P, 4096, 1024, 1024);
    PACKW(Wih2, Wih2P, 4096, 1024, 1024);
    PACKW(Whh2, Whh2P, 4096, 1024, 1024);
    PACKW(Wq,   WqP,   128, 1024, 1024);
    PACKW(Wm,   WmP,   128, 512, 512);
    PACKW(W1,   W1P,   256, 80, 96);
    PACKW(W2,   W2P,   256, 256, 256);
    PACKW(Wmel, WmelP, 80, 1536, 1536);
    PACKW(pw[0], PP[0], 512, 400, 416);
    PACKW(pw[1], PP[1], 512, 2560, 2560);
    PACKW(pw[2], PP[2], 512, 2560, 2560);
    PACKW(pw[3], PP[3], 512, 2560, 2560);
    PACKW(pw[4], PP[4], 80, 2560, 2560);
#undef PACKW

    // ---- zero state (fp32 + bf16) + padded postnet buffers ----
    int nz = 4 * BB * DEC + BB * SS;
    k_zero<<<(nz + 255) / 256, 256, 0, stream>>>(zreg, nz);
    k_zero<<<(BB * DEC + 255) / 256, 256, 0, stream>>>((float*)hBreg, BB * DEC);  // 2*BB*DEC u16
    int nxp = (int)((xp0n + 2 * xpn) / 2);
    k_zero<<<(nxp + 255) / 256, 256, 0, stream>>>((float*)xpall, nxp);

    // ---- hoisted: bf16 encoder copy, teacher-forced input, prenet, processed memory ----
    k_f2b<<<(BB * SS * ENC + 255) / 256, 256, 0, stream>>>(enc, encB, BB * SS * ENC);
    k_din<<<(TT * BB * 96 + 255) / 256, 256, 0, stream>>>(melt, DINB);
    k_prenet1<<<2000, 256, 0, stream>>>(DINB, W1P, b1, P1B);
    k_prenet2<<<2000, 256, 0, stream>>>(P1B, W2P, b2, PB);
    k_pm<<<1024, 256, 0, stream>>>(encB, WmP, pmem);

    // ---- sequential decode ----
    for (int t = 0; t < TT; ++t) {
        const float* prev = (t == 0) ? attnZ : ((t & 1) ? attnA : attnB);
        float* curA = (t & 1) ? attnB : attnA;
        k_recurrent<<<130, 256, 0, stream>>>(h1B, h2B, WqP, Whh1P, Whh2P, q, gh1, gh2);
        k_attention<<<BB, SS, 0, stream>>>(prev, q, pmem, Wconv, Wloc, vvec, enc,
                                           out_attn, curA, ctx, ctxB, t);
        k_lstm<<<64, 256, 0, stream>>>(PB + (size_t)t * BB * PRE, PRE, ctxB, ENC, PRE, 768,
                                       Wih1P, gh1, bih1, bhh1, h1, h1B, c1);
        k_lstm<<<64, 256, 0, stream>>>(h1B, DEC, h1B, DEC, DEC, DEC,
                                       Wih2P, gh2, bih2, bhh2, h2, h2B, c2);
        k_proj<<<1, 384, 0, stream>>>(h2B, ctxB, h2, ctx, WmelP, bmel, Wstop, bstop,
                                      out_mel, out_stop, t);
    }

    // ---- postnet ----
    k_padfill<<<(BB * MEL * TT + 255) / 256, 256, 0, stream>>>(out_mel, Xp0);
    k_postconv<<<4000, 256, 0, stream>>>(Xp0, 84, PP[0], 512, 416,
                                         pb[0], pg[0], pbe[0], pmn[0], pvr[0], 0, nullptr, XpA, nullptr);
    k_postconv<<<4000, 256, 0, stream>>>(XpA, 512, PP[1], 512, 2560,
                                         pb[1], pg[1], pbe[1], pmn[1], pvr[1], 0, nullptr, XpB, nullptr);
    k_postconv<<<4000, 256, 0, stream>>>(XpB, 512, PP[2], 512, 2560,
                                         pb[2], pg[2], pbe[2], pmn[2], pvr[2], 0, nullptr, XpA, nullptr);
    k_postconv<<<4000, 256, 0, stream>>>(XpA, 512, PP[3], 512, 2560,
                                         pb[3], pg[3], pbe[3], pmn[3], pvr[3], 0, nullptr, XpB, nullptr);
    k_postconv<<<625, 256, 0, stream>>>(XpB, 512, PP[4], 80, 2560,
                                        pb[4], nullptr, nullptr, nullptr, nullptr, 1, out_mel, nullptr, out_post);
}
